// ScaledDotProductAttention_42288247996695
// MI455X (gfx1250) — matrix operand order repaired and checked by reading, not compiled, not run on hardware
//
#include <hip/hip_runtime.h>

// Problem constants (B=8, H=16, S=1024, D=64, fp32 in/out)
#define NBH    128         // B*H
#define SEQ    1024
#define DIM    64
#define QSCALE 0.125f      // 1/sqrt(64)

#define KSTRIDE 72         // bf16 elems per K row in LDS (64 + 8 pad -> conflict-free b128)
#define VSTRIDE 1032       // bf16 elems per V^T row in LDS (1024 + 8 pad)
#define LDS_BYTES ((SEQ*KSTRIDE + DIM*VSTRIDE + 8*16*32) * 2)

typedef __attribute__((ext_vector_type(16))) __bf16 v16bf;
typedef __attribute__((ext_vector_type(8)))  float  v8f;

// round-to-nearest-even f32 -> bf16 (scalar)
__device__ __forceinline__ __bf16 f2bf(float f) {
  unsigned u = __builtin_bit_cast(unsigned, f);
  u += 0x7FFFu + ((u >> 16) & 1u);
  unsigned short h = (unsigned short)(u >> 16);
  return __builtin_bit_cast(__bf16, h);
}
// pack two f32 -> two RNE bf16 in one dword
__device__ __forceinline__ unsigned pack2bf(float lo, float hi) {
  unsigned a = __builtin_bit_cast(unsigned, lo);
  unsigned b = __builtin_bit_cast(unsigned, hi);
  a += 0x7FFFu + ((a >> 16) & 1u);
  b += 0x7FFFu + ((b >> 16) & 1u);
  return (a >> 16) | (b & 0xFFFF0000u);
}

__global__ __launch_bounds__(256)
void sdpa_fused(const float* __restrict__ Q, const float* __restrict__ K,
                const float* __restrict__ V, float* __restrict__ Y,
                float* __restrict__ A)
{
  extern __shared__ __attribute__((aligned(16))) unsigned char smem[];
  __bf16* kls = (__bf16*)smem;             // [SEQ][KSTRIDE]  K  bf16 row-major
  __bf16* vls = kls + SEQ * KSTRIDE;       // [DIM][VSTRIDE]  V^T bf16
  __bf16* pls = vls + DIM * VSTRIDE;       // [8 waves][16*32] P transpose staging

  const int tid  = threadIdx.x;
  const int lane = tid & 31;
  const int wid  = tid >> 5;
  const int l16  = lane & 15;
  const int half = lane >> 4;              // 0: lanes 0-15, 1: lanes 16-31
  const int bh   = blockIdx.x >> 1;
  const int qhal = blockIdx.x & 1;         // which 512-row half of the sequence

  const float* qg = Q + (size_t)bh * SEQ * DIM;
  const float* kg = K + (size_t)bh * SEQ * DIM;
  const float* vg = V + (size_t)bh * SEQ * DIM;
  float*       Yg = Y + (size_t)bh * SEQ * DIM;
  float*       Ag = A + (size_t)bh * SEQ * SEQ;
  __bf16*     myp = pls + wid * (16 * 32);

  // ---------------- cooperative staging: K -> bf16 LDS (row-major)
#pragma unroll 1
  for (int it = 0; it < 128; ++it) {
    int idx = it * 256 + tid;              // [0, 32768) : 32 dword-pairs per row
    int kr = idx >> 5, dp = (idx & 31) << 1;
    float2 f = *(const float2*)(kg + kr * DIM + dp);
    *(unsigned*)(kls + kr * KSTRIDE + dp) = pack2bf(f.x, f.y);
  }
  // ---------------- cooperative staging: V -> bf16 LDS, transposed [d][k]
#pragma unroll 1
  for (int it = 0; it < 128; ++it) {
    int idx = it * 256 + tid;              // [0, 32768) : d x key-pair
    int d = idx & 63, kp = idx >> 6;
    float a = vg[(size_t)(2 * kp) * DIM + d];
    float b = vg[(size_t)(2 * kp + 1) * DIM + d];
    *(unsigned*)(vls + d * VSTRIDE + 2 * kp) = pack2bf(a, b);
  }
  __syncthreads();

  // ---------------- 4 q-chunks of 128 rows each (16 rows per wave)
#pragma unroll 1
  for (int qc = 0; qc < 4; ++qc) {
    const int m0 = qhal * 512 + qc * 128 + wid * 16;

    // Q tile as two 16x32 bf16 A-operands (scale folded).
    // A-operand layout: elem j -> K = j + 16*half   (+32*c)
    const float* qrow = qg + (size_t)(m0 + l16) * DIM;
    v16bf qa[2];
#pragma unroll
    for (int c = 0; c < 2; ++c)
#pragma unroll
      for (int j = 0; j < 16; ++j) {
        int k = j + 16 * half + 32 * c;
        qa[c][j] = f2bf(qrow[k] * QSCALE);
      }

    // -------- pass 1: per-lane online max/sum over all 64 key tiles (no shuffles)
    float lmax[8], lsum[8];
#pragma unroll
    for (int i = 0; i < 8; ++i) { lmax[i] = -3.4e38f; lsum[i] = 0.f; }

#pragma unroll 1
    for (int nt = 0; nt < 64; ++nt) {
      const __bf16* krow = kls + (nt * 16 + l16) * KSTRIDE;
      v8f acc = {};
#pragma unroll
      for (int c = 0; c < 2; ++c) {
        // B-operand (K^T): elem j -> d = j + 16*half + 32*c  (contiguous 32B)
        union { v16bf bf; uint4 u[2]; } kb;
        const uint4* p = (const uint4*)(krow + 32 * c + 16 * half);
        kb.u[0] = p[0]; kb.u[1] = p[1];
        acc = __builtin_amdgcn_wmma_f32_16x16x32_bf16(false, qa[c], false, kb.bf,
                                                      (short)0, acc, false, false);
      }
#pragma unroll
      for (int i = 0; i < 8; ++i) {
        float x = acc[i];
        float m = fmaxf(lmax[i], x);
        lsum[i] = lsum[i] * __expf(lmax[i] - m) + __expf(x - m);
        lmax[i] = m;
      }
    }
    // single cross-lane merge of (max,sum) across the 16 lanes of each row
#pragma unroll
    for (int s = 1; s <= 8; s <<= 1) {
#pragma unroll
      for (int i = 0; i < 8; ++i) {
        float om = __shfl_xor(lmax[i], s, 32);
        float os = __shfl_xor(lsum[i], s, 32);
        float nm = fmaxf(lmax[i], om);
        lsum[i] = lsum[i] * __expf(lmax[i] - nm) + os * __expf(om - nm);
        lmax[i] = nm;
      }
    }
#pragma unroll
    for (int i = 0; i < 8; ++i) lsum[i] = 1.f / lsum[i];   // now inverse denom

    // -------- pass 2: recompute S, write normalized A, accumulate Y = P@V
    v8f yacc[4] = {};
#pragma unroll 1
    for (int np = 0; np < 32; ++np) {
#pragma unroll
      for (int t = 0; t < 2; ++t) {
        const int nt = np * 2 + t;
        const __bf16* krow = kls + (nt * 16 + l16) * KSTRIDE;
        v8f acc = {};
#pragma unroll
        for (int c = 0; c < 2; ++c) {
          union { v16bf bf; uint4 u[2]; } kb;
          const uint4* p = (const uint4*)(krow + 32 * c + 16 * half);
          kb.u[0] = p[0]; kb.u[1] = p[1];
          acc = __builtin_amdgcn_wmma_f32_16x16x32_bf16(false, qa[c], false, kb.bf,
                                                        (short)0, acc, false, false);
        }
#pragma unroll
        for (int i = 0; i < 8; ++i) {
          float pv = __expf(acc[i] - lmax[i]) * lsum[i];
          int row = i + 8 * half;                          // C/D layout row
          Ag[(size_t)(m0 + row) * SEQ + nt * 16 + l16] = pv;
          myp[row * 32 + t * 16 + l16] = f2bf(pv);
        }
      }
      asm volatile("s_wait_dscnt 0x0" ::: "memory");       // wave-local LDS ordering

      // P back as A-operand: lane = row l16, two contiguous 16B chunks
      v16bf pa;
      {
        union { v16bf bf; uint4 u[2]; } pu;
        pu.u[0] = *(const uint4*)(myp + l16 * 32 + half * 16);
        pu.u[1] = *(const uint4*)(myp + l16 * 32 + half * 16 + 8);
        pa = pu.bf;
      }

      // V chunk from transposed LDS: B-operand elem j -> k = np*32 + j + 16*half
#pragma unroll
      for (int dt = 0; dt < 4; ++dt) {
        union { v16bf bf; uint4 u[2]; } vb;
        const uint4* p = (const uint4*)(vls + (dt * 16 + l16) * VSTRIDE + np * 32 + 16 * half);
        vb.u[0] = p[0]; vb.u[1] = p[1];
        yacc[dt] = __builtin_amdgcn_wmma_f32_16x16x32_bf16(false, pa, false, vb.bf,
                                                           (short)0, yacc[dt], false, false);
      }
    }

    // -------- store Y (P already normalized)
#pragma unroll
    for (int dt = 0; dt < 4; ++dt)
#pragma unroll
      for (int i = 0; i < 8; ++i) {
        int row = i + 8 * half;
        Yg[(size_t)(m0 + row) * DIM + dt * 16 + l16] = yacc[dt][i];
      }
  }
}

extern "C" void kernel_launch(void* const* d_in, const int* in_sizes, int n_in,
                              void* d_out, int out_size, void* d_ws, size_t ws_size,
                              hipStream_t stream) {
  const float* Q = (const float*)d_in[0];
  const float* K = (const float*)d_in[1];
  const float* V = (const float*)d_in[2];
  float* Y = (float*)d_out;
  float* A = Y + (size_t)NBH * SEQ * DIM;   // outputs concatenated: Y then A

  dim3 grid(NBH * 2);                       // 256 workgroups, 2 per (b,h)
  dim3 block(256);                          // 8 waves (wave32)
  sdpa_fused<<<grid, block, LDS_BYTES, stream>>>(Q, K, V, Y, A);
}
